// Sage_42322607735200
// MI455X (gfx1250) — matrix operand order repaired and checked by reading, not compiled, not run on hardware
//
#include <hip/hip_runtime.h>
#include <hip/hip_bf16.h>

// GraphSAGE 2-layer forward for MI455X (gfx1250, wave32, WMMA bf16).
//   deg -> scatter-mean -> [mean|x]@[W1l;W1r] (WMMA, K=256) -> ReLU -> h(bf16)
//   -> scatter-mean -> [mean2|h]@[W2l;W2r] (WMMA, K=512) -> log_softmax.
// All operands are pre-converted to bf16 in WMMA-fragment-friendly layouts so
// the GEMM inner loop is pure b128 loads + v_wmma.

#define F_IN 128
#define HID  256
#define CLS  40
#define CLSP 48   // CLS padded to 3 col tiles

typedef __bf16 bf16_t;
typedef __attribute__((ext_vector_type(16))) bf16_t v16bf;
typedef __attribute__((ext_vector_type(8)))  bf16_t v8bf;
typedef __attribute__((ext_vector_type(8)))  float  v8f;

__device__ __forceinline__ unsigned short bfbits(float f) {
  union { bf16_t b; unsigned short u; } x; x.b = (bf16_t)f; return x.u;
}
__device__ __forceinline__ float bits2f(unsigned short u) {
  union { unsigned u; float f; } x; x.u = ((unsigned)u) << 16; return x.f;
}
__device__ __forceinline__ unsigned pk2(float lo, float hi) {
  return (unsigned)bfbits(lo) | ((unsigned)bfbits(hi) << 16);
}
__device__ __forceinline__ v8bf ld8(const unsigned short* p) {
  return *(const v8bf*)p;
}
__device__ __forceinline__ v16bf cat16(v8bf lo, v8bf hi) {
  return __builtin_shufflevector(lo, hi, 0,1,2,3,4,5,6,7,8,9,10,11,12,13,14,15);
}

__global__ void k_zero(float* __restrict__ p, long long n) {
  long long i = (long long)blockIdx.x * blockDim.x + threadIdx.x;
  long long stride = (long long)gridDim.x * blockDim.x;
  for (; i < n; i += stride) p[i] = 0.0f;
}

// ---- weight prep: transposed, concatenated, bf16 ----------------------------
// w1t[col][k], col<256, k<256:  k<128 -> W1l[k][col], else W1r[k-128][col]
__global__ void k_prep_w1(const float* __restrict__ W1l, const float* __restrict__ W1r,
                          unsigned short* __restrict__ w1t) {
  int idx = blockIdx.x * blockDim.x + threadIdx.x;   // col*256 + k
  if (idx >= HID * 256) return;
  int ncol = idx >> 8;
  int k    = idx & 255;
  float v = (k < F_IN) ? W1l[k * HID + ncol] : W1r[(k - F_IN) * HID + ncol];
  w1t[idx] = bfbits(v);
}
// w2t[col][k], col<48, k<512: col<40 ? (k<256 ? W2l[k][col] : W2r[k-256][col]) : 0
__global__ void k_prep_w2(const float* __restrict__ W2l, const float* __restrict__ W2r,
                          unsigned short* __restrict__ w2t) {
  int idx = blockIdx.x * blockDim.x + threadIdx.x;   // col*512 + k
  if (idx >= CLSP * 512) return;
  int ncol = idx >> 9;
  int k    = idx & 511;
  float v = 0.0f;
  if (ncol < CLS) v = (k < HID) ? W2l[k * CLS + ncol] : W2r[(k - HID) * CLS + ncol];
  w2t[idx] = bfbits(v);
}

// ---- graph aggregation ------------------------------------------------------
__global__ void k_deg(const int* __restrict__ dst, float* __restrict__ cnt, int E) {
  int e = blockIdx.x * blockDim.x + threadIdx.x;
  if (e < E) unsafeAtomicAdd(&cnt[dst[e]], 1.0f);
}

__global__ void k_rcp(float* __restrict__ cnt, int n) {
  int i = blockIdx.x * blockDim.x + threadIdx.x;
  if (i < n) cnt[i] = 1.0f / fmaxf(cnt[i], 1.0f);
}

// agg1[dst] += x[src], 4 floats per thread (32 chunks per edge).
__global__ void k_scatter1(const float4* __restrict__ x4, const int* __restrict__ src,
                           const int* __restrict__ dst, float* __restrict__ agg1, int E) {
  long long t = (long long)blockIdx.x * blockDim.x + threadIdx.x;
  int e = (int)(t >> 5);
  int q = (int)(t & 31);
  if (e >= E) return;
  int s = src[e], d = dst[e];
  float4 v = x4[(long long)s * (F_IN / 4) + q];
  float* p = agg1 + (long long)d * F_IN + q * 4;
  unsafeAtomicAdd(p + 0, v.x);
  unsafeAtomicAdd(p + 1, v.y);
  unsafeAtomicAdd(p + 2, v.z);
  unsafeAtomicAdd(p + 3, v.w);
}

// agg2[dst] += h[src] (bf16 gather, f32 atomic accumulate), 8 elems per thread.
__global__ void k_scatter2(const unsigned short* __restrict__ hbf, const int* __restrict__ src,
                           const int* __restrict__ dst, float* __restrict__ agg2, int E) {
  long long t = (long long)blockIdx.x * blockDim.x + threadIdx.x;
  int e = (int)(t >> 5);
  int q = (int)(t & 31);
  if (e >= E) return;
  int s = src[e], d = dst[e];
  uint4 raw = *(const uint4*)(hbf + (long long)s * HID + q * 8);
  float* p = agg2 + (long long)d * HID + q * 8;
  unsafeAtomicAdd(p + 0, bits2f((unsigned short)(raw.x & 0xffffu)));
  unsafeAtomicAdd(p + 1, bits2f((unsigned short)(raw.x >> 16)));
  unsafeAtomicAdd(p + 2, bits2f((unsigned short)(raw.y & 0xffffu)));
  unsafeAtomicAdd(p + 3, bits2f((unsigned short)(raw.y >> 16)));
  unsafeAtomicAdd(p + 4, bits2f((unsigned short)(raw.z & 0xffffu)));
  unsafeAtomicAdd(p + 5, bits2f((unsigned short)(raw.z >> 16)));
  unsafeAtomicAdd(p + 6, bits2f((unsigned short)(raw.w & 0xffffu)));
  unsafeAtomicAdd(p + 7, bits2f((unsigned short)(raw.w >> 16)));
}

// ---- A-operand prep ---------------------------------------------------------
// acat[row][0:128)  = bf16(agg1[row]*rc[row]) ; acat[row][128:256) = bf16(x[row])
__global__ void k_build_a1(const float4* __restrict__ agg1v, const float4* __restrict__ xv,
                           const float* __restrict__ rc, unsigned short* __restrict__ acat,
                           long long n) {
  long long t = (long long)blockIdx.x * blockDim.x + threadIdx.x;
  if (t >= n * 32) return;
  long long row = t >> 5;
  int q = (int)(t & 31);                 // feature chunk: 4 floats
  float r = rc[row];
  float4 a = agg1v[row * 32 + q];
  float4 xx = xv[row * 32 + q];
  unsigned short* p = acat + row * HID + q * 4;
  uint2 w0; w0.x = pk2(a.x * r, a.y * r); w0.y = pk2(a.z * r, a.w * r);
  *(uint2*)p = w0;
  uint2 w1; w1.x = pk2(xx.x, xx.y);      w1.y = pk2(xx.z, xx.w);
  *(uint2*)(p + F_IN) = w1;
}

// acat[row][0:256) = bf16(agg2[row]*rc[row])   (reuses the layer-1 buffer)
__global__ void k_build_a2(const float4* __restrict__ agg2v, const float* __restrict__ rc,
                           unsigned short* __restrict__ acat, long long n) {
  long long t = (long long)blockIdx.x * blockDim.x + threadIdx.x;
  if (t >= n * 64) return;
  long long row = t >> 6;
  int q = (int)(t & 63);
  float r = rc[row];
  float4 a = agg2v[row * 64 + q];
  uint2 w; w.x = pk2(a.x * r, a.y * r); w.y = pk2(a.z * r, a.w * r);
  *(uint2*)(acat + row * HID + q * 4) = w;
}

// ---- GEMM 1: h = relu(acat @ w1t^T + b1), K=256, out bf16 -------------------
// block = 512 threads = 16 waves; block -> 16 rows, wave -> one 16-col tile.
__global__ void __launch_bounds__(512) k_gemm1(
    const unsigned short* __restrict__ acat, const unsigned short* __restrict__ w1t,
    const float* __restrict__ b1v, unsigned short* __restrict__ hbf, int n) {
  const int lane = threadIdx.x & 31;
  const int wave = threadIdx.x >> 5;      // col tile 0..15
  const int m    = lane & 15;
  const int half = lane >> 4;
  const int rowTile = blockIdx.x;
  int row  = rowTile * 16 + m;
  int rowc = row < n ? row : n - 1;
  const unsigned short* arow = acat + (long long)rowc * HID;
  const int ncol = wave * 16 + m;
  const unsigned short* brow = w1t + ncol * 256;
  v8f c = {0.f, 0.f, 0.f, 0.f, 0.f, 0.f, 0.f, 0.f};
#pragma unroll
  for (int kk = 0; kk < 8; ++kk) {
    const int kbase = kk * 32;
    v16bf a = cat16(ld8(arow + kbase + half * 8),
                    ld8(arow + kbase + 16 + half * 8));
    v16bf b = cat16(ld8(brow + kbase + half * 8),
                    ld8(brow + kbase + 16 + half * 8));
    c = __builtin_amdgcn_wmma_f32_16x16x32_bf16(false, a, false, b, (short)0, c, false, false);
  }
  const float bias = b1v[ncol];
#pragma unroll
  for (int v = 0; v < 8; ++v) {
    int r = rowTile * 16 + v + 8 * half;   // C/D layout: VGPR v -> M = v + 8*half
    if (r < n) hbf[(long long)r * HID + ncol] = bfbits(fmaxf(c[v] + bias, 0.0f));
  }
}

// ---- GEMM 2: logits = [acat|h] @ w2t^T + b2, K=512, N=40 (3 masked tiles) ---
// block = 96 threads = 3 waves.
__global__ void __launch_bounds__(96) k_gemm2(
    const unsigned short* __restrict__ acat, const unsigned short* __restrict__ hbf,
    const unsigned short* __restrict__ w2t, const float* __restrict__ b2v,
    float* __restrict__ out, int n) {
  const int lane = threadIdx.x & 31;
  const int wave = threadIdx.x >> 5;      // col tile 0..2
  const int m    = lane & 15;
  const int half = lane >> 4;
  const int rowTile = blockIdx.x;
  int row  = rowTile * 16 + m;
  int rowc = row < n ? row : n - 1;
  const unsigned short* arow0 = acat + (long long)rowc * HID;  // mean2 (K 0..255)
  const unsigned short* arow1 = hbf  + (long long)rowc * HID;  // h     (K 256..511)
  const int ncol = wave * 16 + m;        // 0..47
  const bool nok = ncol < CLS;
  const unsigned short* brow = w2t + ncol * 512;
  v8f c = {0.f, 0.f, 0.f, 0.f, 0.f, 0.f, 0.f, 0.f};
#pragma unroll
  for (int kk = 0; kk < 16; ++kk) {
    const int kbase = kk * 32;
    const unsigned short* ar = (kbase < HID) ? (arow0 + kbase) : (arow1 + (kbase - HID));
    v16bf a = cat16(ld8(ar + half * 8), ld8(ar + 16 + half * 8));
    v16bf b = cat16(ld8(brow + kbase + half * 8),
                    ld8(brow + kbase + 16 + half * 8));
    c = __builtin_amdgcn_wmma_f32_16x16x32_bf16(false, a, false, b, (short)0, c, false, false);
  }
  const float bias = nok ? b2v[ncol] : 0.0f;
#pragma unroll
  for (int v = 0; v < 8; ++v) {
    int r = rowTile * 16 + v + 8 * half;
    if (r < n && nok) out[(long long)r * CLS + ncol] = c[v] + bias;
  }
}

__global__ void k_lsm(float* __restrict__ out, int n) {
  int row = blockIdx.x * blockDim.x + threadIdx.x;
  if (row >= n) return;
  float* p = out + (long long)row * CLS;
  float v[CLS];
  float mx = -3.4e38f;
#pragma unroll
  for (int j = 0; j < CLS; ++j) { v[j] = p[j]; mx = fmaxf(mx, v[j]); }
  float s = 0.0f;
#pragma unroll
  for (int j = 0; j < CLS; ++j) s += expf(v[j] - mx);
  float ls = logf(s);
#pragma unroll
  for (int j = 0; j < CLS; ++j) p[j] = v[j] - mx - ls;
}

extern "C" void kernel_launch(void* const* d_in, const int* in_sizes, int n_in,
                              void* d_out, int out_size, void* d_ws, size_t ws_size,
                              hipStream_t stream) {
  const float* x   = (const float*)d_in[0];
  const int*   ei  = (const int*)d_in[1];
  const float* W1l = (const float*)d_in[2];
  const float* W1r = (const float*)d_in[3];
  const float* b1  = (const float*)d_in[4];
  const float* W2l = (const float*)d_in[5];
  const float* W2r = (const float*)d_in[6];
  const float* b2  = (const float*)d_in[7];
  float* out = (float*)d_out;

  const int n = in_sizes[0] / F_IN;
  const int E = in_sizes[1] / 2;
  const int* src = ei;
  const int* dst = ei + E;

  // Workspace carve-up:
  //   cnt/rc [npad] f32 | agg1 [n*128] f32 | agg2 [n*256] f32
  //   | hbf [n*256] bf16 | acat [n*256] bf16 (shared by layer1/layer2)
  //   | w1t [256*256] bf16 | w2t [48*512] bf16
  const long long npad = ((long long)n + 127) & ~127LL;
  float* cnt  = (float*)d_ws;
  float* agg1 = cnt + npad;
  float* agg2 = agg1 + (long long)n * F_IN;
  unsigned short* hbf  = (unsigned short*)(agg2 + (long long)n * HID);
  unsigned short* acat = hbf + (long long)n * HID;
  unsigned short* w1t  = acat + (long long)n * HID;
  unsigned short* w2t  = w1t + HID * 256;

  const long long zeroN = npad + (long long)n * F_IN + (long long)n * HID;
  k_zero<<<4096, 256, 0, stream>>>(cnt, zeroN);

  k_prep_w1<<<(HID * 256 + 255) / 256, 256, 0, stream>>>(W1l, W1r, w1t);
  k_prep_w2<<<(CLSP * 512 + 255) / 256, 256, 0, stream>>>(W2l, W2r, w2t);

  k_deg<<<(E + 255) / 256, 256, 0, stream>>>(dst, cnt, E);
  k_rcp<<<(n + 255) / 256, 256, 0, stream>>>(cnt, n);

  const long long t32 = (long long)E * 32;
  k_scatter1<<<(unsigned)((t32 + 255) / 256), 256, 0, stream>>>(
      (const float4*)x, src, dst, agg1, E);

  k_build_a1<<<(unsigned)(((long long)n * 32 + 255) / 256), 256, 0, stream>>>(
      (const float4*)agg1, (const float4*)x, cnt, acat, n);

  const int rowTiles = (n + 15) / 16;
  k_gemm1<<<rowTiles, 512, 0, stream>>>(acat, w1t, b1, hbf, n);

  k_scatter2<<<(unsigned)((t32 + 255) / 256), 256, 0, stream>>>(hbf, src, dst, agg2, E);

  k_build_a2<<<(unsigned)(((long long)n * 64 + 255) / 256), 256, 0, stream>>>(
      (const float4*)agg2, cnt, acat, n);

  k_gemm2<<<rowTiles, 96, 0, stream>>>(acat, hbf, w2t, b2, out, n);

  k_lsm<<<(n + 127) / 128, 128, 0, stream>>>(out, n);
}
